// LSTMCell_65618510348829
// MI455X (gfx1250) — compile-verified, matrix operand order checked by reading, not run on hardware
//
#include <hip/hip_runtime.h>
#include <hip/hip_bf16.h>

// ---------------------------------------------------------------------------
// LSTM cell + LayerNorm for MI455X (gfx1250).
// bf16 WMMA (f32 accumulate) + async global->LDS double-buffered pipeline.
// B=16384, I=H=512, K=I+H=1024, gate cols = 4H = 2048.
// Memory-bound (~170MB min traffic @ 23.3 TB/s). LDS staging gives 4x B reuse;
// async copies (ASYNCcnt) decouple global latency from the WMMA pipe; B
// fragments are burst-loaded into registers so the 8 WMMAs per K-chunk issue
// behind a single dscnt wait.
// ---------------------------------------------------------------------------

typedef __attribute__((ext_vector_type(16))) __bf16 v16bf;
typedef __attribute__((ext_vector_type(8)))  __bf16 v8bf;
typedef __attribute__((ext_vector_type(8)))  float  v8f;

#define B_ROWS 16384
#define IN_DIM 512
#define H_DIM  512
#define K_DIM  1024
#define GCOLS  2048
#define WPACK_BYTES (4u << 20)   // 128 ct * 32 kc * 32 lanes * 16 elems * 2B

// LDS layout per buffer: [0,16KB) = 16 B-tiles (1KB each), [16KB,20KB) = 4 A-tiles
#define LDS_BUF_STRIDE 20480
#define LDS_A_OFF      16384

__device__ __forceinline__ float sigmoid_f(float x) {
  return 1.0f / (1.0f + __expf(-x));
}
__device__ __forceinline__ float tanh_f(float x) {
  float a = fabsf(x);
  float e = __expf(-2.0f * a);
  float t = (1.0f - e) / (1.0f + e);
  return copysignf(t, x);
}

// Async copy: 32 lanes x 16B, global -> LDS, tracked by ASYNCcnt.
// vdst VGPR = wave-relative LDS byte offset (low 32 bits of generic pointer).
__device__ __forceinline__ void async_copy_b128(unsigned lds_off, const void* gsrc) {
  asm volatile("global_load_async_to_lds_b128 %0, %1, off"
               :: "v"(lds_off), "v"(gsrc) : "memory");
}

// ---------------------------------------------------------------------------
// Prep 1: pack A = [x | h_prev] as bf16, row-major [B_ROWS][K_DIM].
// ---------------------------------------------------------------------------
__global__ void pack_a_kernel(const float* __restrict__ x,
                              const float* __restrict__ h_prev,
                              __bf16* __restrict__ abf) {
  size_t t   = (size_t)blockIdx.x * blockDim.x + threadIdx.x;
  size_t idx = t * 8;
  if (idx >= (size_t)B_ROWS * K_DIM) return;
  size_t r = idx / K_DIM;
  int    k = (int)(idx % K_DIM);
  const float* src = (k < IN_DIM) ? (x + r * IN_DIM + k)
                                  : (h_prev + r * IN_DIM + (k - IN_DIM));
  float4 f0 = *(const float4*)(src);
  float4 f1 = *(const float4*)(src + 4);
  v8bf o;
  o[0] = (__bf16)f0.x; o[1] = (__bf16)f0.y; o[2] = (__bf16)f0.z; o[3] = (__bf16)f0.w;
  o[4] = (__bf16)f1.x; o[5] = (__bf16)f1.y; o[6] = (__bf16)f1.z; o[7] = (__bf16)f1.w;
  *(v8bf*)(abf + idx) = o;
}

// ---------------------------------------------------------------------------
// Prep 2: pre-swizzle [W_i; W_h] (K_DIM x GCOLS) into per-lane B-fragment
// order for v_wmma_f32_16x16x32_bf16 (mirror of documented 16-bit A layout):
//   lanes 0-15 : N = lane,    K in {0..7, 16..23}
//   lanes 16-31: N = lane-16, K in {8..15, 24..31}
// Blob index: [ct(128)][kc(32)][lane(32)][e(16)] bf16 -> 1KB per (ct,kc).
// ---------------------------------------------------------------------------
__global__ void pack_w_kernel(const float* __restrict__ Wi,
                              const float* __restrict__ Wh,
                              __bf16* __restrict__ wp) {
  unsigned t = blockIdx.x * blockDim.x + threadIdx.x;   // 2,097,152 total
  int e    = t & 15;
  int lane = (t >> 4) & 31;
  int kc   = (t >> 9) & 31;
  int ct   = t >> 14;                                   // 0..127
  int n    = lane & 15;
  int klocal = (lane < 16) ? ((e < 8) ? e : e + 8)
                           : ((e < 8) ? e + 8 : e + 16);
  int kk  = kc * 32 + klocal;                           // 0..1023
  int col = ct * 16 + n;                                // 0..2047
  float v = (kk < IN_DIM) ? Wi[(size_t)kk * GCOLS + col]
                          : Wh[(size_t)(kk - IN_DIM) * GCOLS + col];
  wp[t] = (__bf16)v;
}

// ---------------------------------------------------------------------------
// Main fused kernel: 256 threads = 8 waves; WG tile = 64 rows x 64 h-cols.
// Wave w: row tile rt = w%4 (16 rows), col group cg = w/4 (2 h-tiles of 16).
// Per 32-wide K chunk: 5 async b128 copies per wave stage 16 B-tiles (shared
// by the 4 row-tile waves) + 4 A-tiles into LDS (double buffered). Compute:
// 18 ds_load_b128 -> one dscnt wait -> burst of 8 WMMAs. Epilogue fuses the
// LSTM elementwise math; gates never touch global memory.
// ---------------------------------------------------------------------------
__global__ __launch_bounds__(256)
void lstm_gemm_kernel(const __bf16* __restrict__ abf,
                      const __bf16* __restrict__ wp,
                      const float*  __restrict__ c_prev,
                      const float*  __restrict__ bias,
                      float* __restrict__ h_raw,
                      float* __restrict__ c_out) {
  __shared__ alignas(32) char smem[2 * LDS_BUF_STRIDE];   // 40 KB

  const int lane = threadIdx.x & 31;
  const int w    = threadIdx.x >> 5;
  const int rt   = w & 3;
  const int cg   = w >> 2;
  const int rowblock = blockIdx.y * 64;          // 64 rows per WG
  const int rowbase  = rowblock + rt * 16;
  const int bx4  = blockIdx.x * 4;               // first h-col tile of this WG
  const int n    = lane & 15;
  const int half = lane >> 4;

  int hbase[2];
  hbase[0] = blockIdx.x * 64 + cg * 32;
  hbase[1] = hbase[0] + 16;

  // bias-initialized accumulators (bias is per-column)
  v8f acc[2][4];
#pragma unroll
  for (int t = 0; t < 2; ++t)
#pragma unroll
    for (int g = 0; g < 4; ++g) {
      float bv = bias[g * H_DIM + hbase[t] + n];
      v8f a8;
#pragma unroll
      for (int r = 0; r < 8; ++r) a8[r] = bv;
      acc[t][g] = a8;
    }

  const char* wpB  = (const char*)wp;
  const char* abfB = (const char*)abf;

  // Issue the 5 async b128 copies this wave owns for K-chunk kc into buffer.
  auto issue_chunk = [&](int kc, int buf) {
    const unsigned base = (unsigned)(size_t)smem + buf * LDS_BUF_STRIDE;
    // B: wave w copies WG-tiles i = 2w, 2w+1 (each 1KB = 2 ops).
#pragma unroll
    for (int q = 0; q < 4; ++q) {
      int i = 2 * w + (q >> 1);                 // 0..15
      int p = q & 1;
      int g = i >> 2, j = i & 3;
      int ctile = g * 32 + bx4 + j;             // packed-weight column tile
      const void* gsrc = wpB + (size_t)(ctile * 32 + kc) * 1024 + lane * 32 + p * 16;
      async_copy_b128(base + (unsigned)(i * 1024 + lane * 32 + p * 16), gsrc);
    }
    // A: op w of 8 -> row tile w>>1, half w&1. Gathers row-major A into
    // per-lane fragment order (row = base+lane%16, K half by lane/16).
    {
      int rta = w >> 1, p = w & 1;
      const void* gsrc = abfB +
          ((size_t)(rowblock + rta * 16 + (lane & 15)) * K_DIM +
           (size_t)kc * 32 + (size_t)(lane >> 4) * 8) * 2 + p * 32;
      async_copy_b128(base + (unsigned)(LDS_A_OFF + rta * 1024 + lane * 32 + p * 16), gsrc);
    }
  };

  // Prefetch the epilogue c_prev tile into near caches while we compute.
  __builtin_prefetch(c_prev + (size_t)rowbase * H_DIM + hbase[0], 0, 3);

  issue_chunk(0, 0);

  for (int kc = 0; kc < 32; ++kc) {
    const int cur = kc & 1;
    if (kc + 1 < 32) {
      issue_chunk(kc + 1, cur ^ 1);
      asm volatile("s_wait_asynccnt 0x5" ::: "memory");   // chunk kc's 5 done
    } else {
      asm volatile("s_wait_asynccnt 0x0" ::: "memory");
    }
    __syncthreads();                                      // all waves' copies visible

    const char* bufp = smem + cur * LDS_BUF_STRIDE;

    // Burst-load A + all 8 B fragments into dedicated registers first so the
    // scheduler can issue every ds_load before any WMMA consumes one.
    v16bf afrag = *(const v16bf*)(bufp + LDS_A_OFF + rt * 1024 + lane * 32);
    v16bf bfr[8];
#pragma unroll
    for (int t = 0; t < 2; ++t)
#pragma unroll
      for (int g = 0; g < 4; ++g)
        bfr[t * 4 + g] =
            *(const v16bf*)(bufp + (g * 4 + cg * 2 + t) * 1024 + lane * 32);

#pragma unroll
    for (int t = 0; t < 2; ++t)
#pragma unroll
      for (int g = 0; g < 4; ++g)
        acc[t][g] = __builtin_amdgcn_wmma_f32_16x16x32_bf16(
            /*neg_a=*/false, afrag, /*neg_b=*/false, bfr[t * 4 + g],
            /*c_mod=*/(short)0, acc[t][g],
            /*reuse_a=*/false, /*reuse_b=*/false);

    __syncthreads();     // buffer may be overwritten by next iteration's issue
  }

  // Epilogue: C/D layout -> VGPR r, lanes 0-15: (row rowbase+r, col n),
  // lanes 16-31: (row rowbase+r+8, col n).
#pragma unroll
  for (int t = 0; t < 2; ++t) {
    const int col = hbase[t] + n;
#pragma unroll
    for (int r = 0; r < 8; ++r) {
      const int row = rowbase + r + half * 8;
      const size_t off = (size_t)row * H_DIM + col;
      float iv = sigmoid_f(acc[t][0][r]);
      float fv = sigmoid_f(acc[t][1][r]);
      float gv = tanh_f(acc[t][2][r]);
      float ov = sigmoid_f(acc[t][3][r]);
      float cp = c_prev[off];
      float cv = fv * cp + iv * gv;
      float hv = ov * tanh_f(cv);
      c_out[off] = cv;
      h_raw[off] = hv;   // pre-LayerNorm h, normalized in-place by ln_kernel
    }
  }
}

// ---------------------------------------------------------------------------
// LayerNorm: one wave32 per row (512 elems -> 16 per lane), shuffle-reduce.
// In-place on the h region of d_out (stream-ordered after the GEMM kernel).
// ---------------------------------------------------------------------------
__global__ __launch_bounds__(256)
void ln_kernel(float* __restrict__ h,
               const float* __restrict__ gamma,
               const float* __restrict__ beta) {
  const int lane = threadIdx.x & 31;
  const int w    = threadIdx.x >> 5;
  const int row  = blockIdx.x * 8 + w;
  float* hp = h + (size_t)row * H_DIM;

  float v[16];
  float s = 0.0f, ss = 0.0f;
#pragma unroll
  for (int j = 0; j < 16; ++j) {
    float xv = hp[lane + 32 * j];
    v[j] = xv;
    s  += xv;
    ss += xv * xv;
  }
#pragma unroll
  for (int m = 16; m >= 1; m >>= 1) {
    s  += __shfl_xor(s, m, 32);
    ss += __shfl_xor(ss, m, 32);
  }
  const float mean = s * (1.0f / (float)H_DIM);
  const float var  = ss * (1.0f / (float)H_DIM) - mean * mean;
  const float rstd = rsqrtf(var + 1e-5f);
#pragma unroll
  for (int j = 0; j < 16; ++j) {
    int col = lane + 32 * j;
    hp[col] = (v[j] - mean) * rstd * gamma[col] + beta[col];
  }
}

// ---------------------------------------------------------------------------
extern "C" void kernel_launch(void* const* d_in, const int* in_sizes, int n_in,
                              void* d_out, int out_size, void* d_ws, size_t ws_size,
                              hipStream_t stream) {
  const float* x      = (const float*)d_in[0];
  const float* h_prev = (const float*)d_in[1];
  const float* c_prev = (const float*)d_in[2];
  const float* W_i    = (const float*)d_in[3];
  const float* W_h    = (const float*)d_in[4];
  const float* b      = (const float*)d_in[5];
  const float* ln_w   = (const float*)d_in[6];
  const float* ln_b   = (const float*)d_in[7];

  float* h_out = (float*)d_out;                        // [B_ROWS * H_DIM]
  float* c_out = h_out + (size_t)B_ROWS * H_DIM;       // [B_ROWS * H_DIM]

  __bf16* wp  = (__bf16*)d_ws;                                  // 4 MB packed weights
  __bf16* abf = (__bf16*)((char*)d_ws + WPACK_BYTES);           // 32 MB packed A

  pack_w_kernel<<<8192, 256, 0, stream>>>(W_i, W_h, wp);
  pack_a_kernel<<<8192, 256, 0, stream>>>(x, h_prev, abf);

  // grid.x = 512/64 = 8 h-col blocks, grid.y = 16384/64 = 256 row blocks
  dim3 grid(8, 256);
  lstm_gemm_kernel<<<grid, 256, 0, stream>>>(abf, wp, c_prev, b, h_out, c_out);

  ln_kernel<<<2048, 256, 0, stream>>>(h_out, ln_w, ln_b);
}
